// GridSampleNet_46119358825238
// MI455X (gfx1250) — matrix operand order repaired and checked by reading, not compiled, not run on hardware
//
#include <hip/hip_runtime.h>
#include <math.h>

// ---------------------------------------------------------------------------
// CDNA5 (gfx1250) implementation of GridSampleNet.
// Convs = bf16 implicit-GEMM on V_WMMA_F32_16X16X32_BF16 (wave32).
// Fast path: NHWC bf16 activations + [O][KH][KW][I] bf16 weights -> every
// WMMA fragment comes from contiguous global_load_b128s, no div/mod and
// (for pad==0 convs) no bounds checks in the K loop. One wave = 16x64 output
// tile (4 accumulators sharing one A fragment). Tiles are mt-fastest so the
// 4 waves of a block read the same B addresses (WGP$ hits, 4x B reuse).
// ---------------------------------------------------------------------------

typedef __attribute__((ext_vector_type(16))) __bf16          v16bf;
typedef __attribute__((ext_vector_type(8)))  float           v8f;
typedef __attribute__((ext_vector_type(8)))  unsigned short  us8;

__device__ __forceinline__ unsigned short f2bf(float f) {
  union { float f; unsigned u; } c; c.f = f;
  unsigned u = c.u;
  unsigned r = (u + 0x7fffu + ((u >> 16) & 1u)) >> 16;  // RNE
  return (unsigned short)r;
}
__device__ __forceinline__ __bf16 bfbits(unsigned short s) {
  union { unsigned short u; __bf16 b; } c; c.u = s; return c.b;
}
__device__ __forceinline__ v16bf mkfrag(us8 lo, us8 hi) {
  union { struct { us8 l; us8 h; } s; v16bf v; } u;
  u.s.l = lo; u.s.h = hi; return u.v;
}

// ---------------------------- small utility kernels ------------------------
__global__ void cvt_bf16_k(const float* __restrict__ x, unsigned short* __restrict__ y,
                           long long n) {
  long long i = (long long)blockIdx.x * blockDim.x + threadIdx.x;
  if (i < n) y[i] = f2bf(x[i]);
}

// OIHW f32 -> [O][KH][KW][I] bf16 (fast-path weight layout)
__global__ void wreorder_k(const float* __restrict__ w, unsigned short* __restrict__ y,
                           int O, int I, int Kk) {
  long long n = (long long)O * I * Kk * Kk;
  long long i = (long long)blockIdx.x * blockDim.x + threadIdx.x;
  if (i >= n) return;
  int kw = (int)(i % Kk); long long t = i / Kk;
  int kh = (int)(t % Kk); t /= Kk;
  int ci = (int)(t % I);  int o = (int)(t / I);
  y[(((long long)o * Kk + kh) * Kk + kw) * I + ci] = f2bf(w[i]);
}

__global__ void fill0_k(float* __restrict__ y, long long n) {
  long long i = (long long)blockIdx.x * blockDim.x + threadIdx.x;
  if (i < n) y[i] = 0.f;
}

__global__ void copy_k(const float* __restrict__ x, float* __restrict__ y, long long n) {
  long long i = (long long)blockIdx.x * blockDim.x + threadIdx.x;
  if (i < n) y[i] = x[i];
}

// NCHW f32 -> NHWC bf16
__global__ void nchw2nhwc_bf_k(const float* __restrict__ x, unsigned short* __restrict__ y,
                               int B, int C, int HW) {
  long long n = (long long)B * C * HW;
  long long i = (long long)blockIdx.x * blockDim.x + threadIdx.x;
  if (i >= n) return;
  int j = (int)(i % HW); long long t = i / HW;
  int c = (int)(t % C);  int b = (int)(t / C);
  y[((long long)b * HW + j) * C + c] = f2bf(x[i]);
}

// src (B,Cs,HW) -> dst (B,Cd,HW) at channel offset coff (both NCHW f32)
__global__ void copy_into_k(const float* __restrict__ src, float* __restrict__ dst,
                            int B, int Cs, int HW, int Cd, int coff) {
  long long n = (long long)B * Cs * HW;
  long long i = (long long)blockIdx.x * blockDim.x + threadIdx.x;
  if (i >= n) return;
  int j = (int)(i % HW); long long t = i / HW;
  int c = (int)(t % Cs); int b = (int)(t / Cs);
  dst[((long long)(b * Cd + coff + c)) * HW + j] = src[i];
}

// -------------------- Sobel / dem-mag / bf16 stem input --------------------
__global__ void sobel_k(const float* __restrict__ x, float* __restrict__ grads,
                        unsigned short* __restrict__ finbf, int B, int H, int W) {
  long long n = (long long)B * H * W;
  long long i = (long long)blockIdx.x * blockDim.x + threadIdx.x;
  if (i >= n) return;
  int w = (int)(i % W); long long t = i / W;
  int h = (int)(t % H); int b = (int)(t / H);
  const float* dem = x + ((long long)(b * 3 + 2)) * H * W;
  auto at = [&](int yy, int xx) -> float {
    yy = yy < 0 ? 0 : (yy > H - 1 ? H - 1 : yy);
    xx = xx < 0 ? 0 : (xx > W - 1 ? W - 1 : xx);
    return dem[(long long)yy * W + xx];
  };
  float gx = (-at(h - 1, w - 1) + at(h - 1, w + 1)
              - 2.f * at(h, w - 1) + 2.f * at(h, w + 1)
              - at(h + 1, w - 1) + at(h + 1, w + 1)) * 0.125f;
  float gy = (-at(h - 1, w - 1) - 2.f * at(h - 1, w) - at(h - 1, w + 1)
              + at(h + 1, w - 1) + 2.f * at(h + 1, w) + at(h + 1, w + 1)) * 0.125f;
  grads[((long long)(b * 2 + 0) * H + h) * W + w] = gx;
  grads[((long long)(b * 2 + 1) * H + h) * W + w] = gy;
  float mag = sqrtf(gx * gx + gy * gy);
  finbf[((long long)(b * 3 + 0) * H + h) * W + w] = f2bf(x[((long long)(b * 3 + 0) * H + h) * W + w]);
  finbf[((long long)(b * 3 + 1) * H + h) * W + w] = f2bf(x[((long long)(b * 3 + 1) * H + h) * W + w]);
  finbf[((long long)(b * 3 + 2) * H + h) * W + w] = f2bf(mag);
}

// ---------------------- WMMA conv: fast path -------------------------------
// Requires Cin % 32 == 0, Cout % 16 == 0. BOUNDED=false iff pad==0 (then every
// (ih,iw) is provably in range and all bounds checks vanish from the K loop).
template <bool BOUNDED>
__global__ __launch_bounds__(128)
void conv_wmma_fast_k(const unsigned short* __restrict__ Wb,  // [Cout][KH][KW][Cin] bf16
                      const unsigned short* __restrict__ Xb,  // NHWC bf16
                      float* __restrict__ Y,                  // NCHW f32
                      int B, int Cin, int H, int W,
                      int Cout, int KH, int KW, int stride, int pad,
                      int OH, int OW) {
  const int lane = threadIdx.x & 31;
  const int wave = threadIdx.x >> 5;
  const int Ntot = B * OH * OW;
  const int tN = (Ntot + 63) >> 6;   // 64 output columns per wave
  const int tM = Cout >> 4;
  int tile = blockIdx.x * (blockDim.x >> 5) + wave;
  const int tmax = tM * tN - 1;
  tile = tile > tmax ? tmax : tile;  // wave-uniform clamp: EXEC all-1s at WMMA
  // mt-fastest: the 4 waves of a block share the same B tile (WGP$ hits).
  const int mt = tile % tM, nt = tile / tM;

  const int akb = (lane < 16) ? 0 : 8;
  const int bkb = (lane < 16) ? 0 : 8;
  const int mrow = mt * 16 + (lane & 15);           // Cout % 16 == 0 -> always valid
  const long long K = (long long)KH * KW * Cin;
  const unsigned short* wrow = Wb + (long long)mrow * K;

  int bb[4], oh[4], ow[4]; bool nv[4];
#pragma unroll
  for (int q = 0; q < 4; ++q) {
    int nc = nt * 64 + q * 16 + (lane & 15);
    nv[q] = nc < Ntot;
    nc = nv[q] ? nc : Ntot - 1;
    ow[q] = nc % OW; int t = nc / OW;
    oh[q] = t % OH;  bb[q] = t / OH;
  }

  v8f acc[4] = {};

  for (int kh = 0; kh < KH; ++kh) {
    for (int kw = 0; kw < KW; ++kw) {
      const unsigned short* wk = wrow + (long long)(kh * KW + kw) * Cin;
      const unsigned short* xp[4]; bool ok[4];
#pragma unroll
      for (int q = 0; q < 4; ++q) {
        int ih = oh[q] * stride - pad + kh;
        int iw = ow[q] * stride - pad + kw;
        ok[q] = !BOUNDED || (((unsigned)ih < (unsigned)H) & ((unsigned)iw < (unsigned)W));
        xp[q] = Xb + (((long long)bb[q] * H + ih) * W + iw) * Cin + bkb;
      }
      for (int ci = 0; ci < Cin; ci += 32) {
        us8 a0 = *(const us8*)(wk + ci + akb);
        us8 a1 = *(const us8*)(wk + ci + akb + 16);
        v16bf av = mkfrag(a0, a1);
#pragma unroll
        for (int q = 0; q < 4; ++q) {
          us8 b0 = {0, 0, 0, 0, 0, 0, 0, 0};
          us8 b1 = {0, 0, 0, 0, 0, 0, 0, 0};
          if (ok[q]) {
            b0 = *(const us8*)(xp[q] + ci);
            b1 = *(const us8*)(xp[q] + ci + 16);
          }
          v16bf bv = mkfrag(b0, b1);
          acc[q] = __builtin_amdgcn_wmma_f32_16x16x32_bf16(false, av, false, bv,
                                                           (short)0, acc[q], false, false);
        }
      }
    }
  }

  // store: decode kept from entry -> no division in epilogue
#pragma unroll
  for (int q = 0; q < 4; ++q) {
    if (nv[q]) {
      long long basep = ((long long)bb[q] * Cout) * OH * OW + (long long)oh[q] * OW + ow[q];
      long long mbase = (long long)(mt * 16 + ((lane >= 16) ? 8 : 0)) * OH * OW;
#pragma unroll
      for (int r = 0; r < 8; ++r)
        Y[basep + mbase + (long long)r * OH * OW] = acc[q][r];
    }
  }
}

// ---------------------- WMMA conv: generic path (stem only) ----------------
__global__ void conv_wmma_gen_k(const unsigned short* __restrict__ Wb,  // [Cout][Cin*KH*KW]
                                const unsigned short* __restrict__ Xb,  // NCHW bf16
                                float* __restrict__ Y,
                                int B, int Cin, int H, int W,
                                int Cout, int KH, int KW, int stride, int pad,
                                int OH, int OW) {
  const int lane = threadIdx.x & 31;
  const int wave = threadIdx.x >> 5;
  const int Ntot = B * OH * OW;
  const int tN = (Ntot + 15) >> 4;
  const int tM = (Cout + 15) >> 4;
  int tile = blockIdx.x * (blockDim.x >> 5) + wave;
  const int tmax = tM * tN - 1;
  tile = tile > tmax ? tmax : tile;
  const int mt = tile / tN, nt = tile % tN;
  const int K = Cin * KH * KW;
  const int khkw = KH * KW;

  const int  mrow = mt * 16 + (lane & 15);
  const bool mval = mrow < Cout;
  const int  akb  = (lane < 16) ? 0 : 8;
  const int  ncol = nt * 16 + (lane & 15);
  const bool nval = ncol < Ntot;
  const int  bkb  = (lane < 16) ? 0 : 8;
  int nc = nval ? ncol : (Ntot - 1);
  const int ow = nc % OW; int t = nc / OW;
  const int oh = t % OH;  const int bb = t / OH;
  const int ih0 = oh * stride - pad, iw0 = ow * stride - pad;

  v8f acc = {};
  for (int k0 = 0; k0 < K; k0 += 32) {
    v16bf a, b;
#pragma unroll
    for (int e = 0; e < 16; ++e) {
      int kk = k0 + akb + ((e >> 3) << 4) + (e & 7);
      unsigned short u = (mval && kk < K) ? Wb[mrow * K + kk] : (unsigned short)0;
      a[e] = bfbits(u);
    }
#pragma unroll
    for (int e = 0; e < 16; ++e) {
      int kk = k0 + bkb + ((e >> 3) << 4) + (e & 7);
      unsigned short u = 0;
      if (nval && kk < K) {
        int ci = kk / khkw; int r = kk - ci * khkw;
        int kh = r / KW,    kw = r - kh * KW;
        int ih = ih0 + kh,  iw = iw0 + kw;
        if ((unsigned)ih < (unsigned)H && (unsigned)iw < (unsigned)W)
          u = Xb[((long long)(bb * Cin + ci) * H + ih) * W + iw];
      }
      b[e] = bfbits(u);
    }
    acc = __builtin_amdgcn_wmma_f32_16x16x32_bf16(false, a, false, b,
                                                  (short)0, acc, false, false);
  }
#pragma unroll
  for (int r = 0; r < 8; ++r) {
    int mm = mt * 16 + r + ((lane >= 16) ? 8 : 0);
    int nn = nt * 16 + (lane & 15);
    if (mm < Cout && nn < Ntot) {
      int oww = nn % OW; int t2 = nn / OW;
      int ohh = t2 % OH; int b2 = t2 / OH;
      Y[((long long)(b2 * Cout + mm) * OH + ohh) * OW + oww] = acc[r];
    }
  }
}

// ------------------------------ batch norm ---------------------------------
__global__ void bn_stats_k(const float* __restrict__ x, float* __restrict__ mean,
                           float* __restrict__ rstd, int B, int C, int HW, float eps) {
  const int c = blockIdx.x;
  float s = 0.f, s2 = 0.f;
  const long long n = (long long)B * HW;
  for (long long i = threadIdx.x; i < n; i += blockDim.x) {
    int b = (int)(i / HW), j = (int)(i % HW);
    float v = x[((long long)(b * C + c)) * HW + j];
    s += v; s2 += v * v;
  }
  for (int off = 16; off; off >>= 1) { s += __shfl_down(s, off); s2 += __shfl_down(s2, off); }
  __shared__ float sh[2][8];
  int lane = threadIdx.x & 31, w = threadIdx.x >> 5;
  if (lane == 0) { sh[0][w] = s; sh[1][w] = s2; }
  __syncthreads();
  if (threadIdx.x == 0) {
    float S = 0.f, S2 = 0.f; int nw = blockDim.x >> 5;
    for (int i = 0; i < nw; ++i) { S += sh[0][i]; S2 += sh[1][i]; }
    float m = S / (float)n;
    float var = S2 / (float)n - m * m;
    mean[c] = m; rstd[c] = rsqrtf(var + eps);
  }
}

// y = act((x-mean)*rstd) (NCHW f32); ybf optional NHWC bf16. act:0 none,1 relu,2 tanh
__global__ void bn_act_k(const float* __restrict__ x, const float* __restrict__ mean,
                         const float* __restrict__ rstd, float* __restrict__ y,
                         unsigned short* __restrict__ ybf, int C, int HW,
                         long long n, int act) {
  long long i = (long long)blockIdx.x * blockDim.x + threadIdx.x;
  if (i >= n) return;
  int j = (int)(i % HW); long long t = i / HW;
  int c = (int)(t % C);  int b = (int)(t / C);
  float v = (x[i] - mean[c]) * rstd[c];
  if (act == 1)      v = fmaxf(v, 0.f);
  else if (act == 2) v = tanhf(v);
  if (y)   y[i] = v;
  if (ybf) ybf[((long long)b * HW + j) * C + c] = f2bf(v);
}

// y = relu(a + res) (NCHW f32); ybf optional NHWC bf16
__global__ void add_relu_k(const float* __restrict__ a, const float* __restrict__ res,
                           float* __restrict__ y, unsigned short* __restrict__ ybf,
                           int C, int HW, long long n) {
  long long i = (long long)blockIdx.x * blockDim.x + threadIdx.x;
  if (i >= n) return;
  float v = fmaxf(a[i] + res[i], 0.f);
  y[i] = v;
  if (ybf) {
    int j = (int)(i % HW); long long t = i / HW;
    int c = (int)(t % C);  int b = (int)(t / C);
    ybf[((long long)b * HW + j) * C + c] = f2bf(v);
  }
}

// ------------------------------ pooling ------------------------------------
__global__ void maxpool_k(const float* __restrict__ x, float* __restrict__ y,
                          unsigned short* __restrict__ ybf, int B, int C, int H, int W) {
  const int OH = H / 2, OW = W / 2;
  long long n = (long long)B * C * OH * OW;
  long long i = (long long)blockIdx.x * blockDim.x + threadIdx.x;
  if (i >= n) return;
  int ow = (int)(i % OW); long long t = i / OW;
  int oh = (int)(t % OH); t /= OH;
  int c = (int)(t % C);   int b = (int)(t / C);
  float m = -3.0e38f;
  for (int kh = 0; kh < 3; ++kh) {
    int ih = oh * 2 - 1 + kh; if ((unsigned)ih >= (unsigned)H) continue;
    for (int kw = 0; kw < 3; ++kw) {
      int iw = ow * 2 - 1 + kw; if ((unsigned)iw >= (unsigned)W) continue;
      float v = x[((long long)(b * C + c) * H + ih) * W + iw];
      m = v > m ? v : m;
    }
  }
  y[i] = m;
  if (ybf) ybf[((long long)b * OH * OW + (long long)oh * OW + ow) * C + c] = f2bf(m);
}

// grads (B,2,H,H) -> grid (B,OH,OH,2): 4x4 avgpool * gs[0] * (OH/64)
__global__ void avg_grid_k(const float* __restrict__ grads, const float* __restrict__ gs,
                           float* __restrict__ grid, int B, int H, int OH) {
  long long n = (long long)B * 2 * OH * OH;
  long long i = (long long)blockIdx.x * blockDim.x + threadIdx.x;
  if (i >= n) return;
  int w = (int)(i % OH); long long t = i / OH;
  int h = (int)(t % OH); t /= OH;
  int c = (int)(t % 2);  int b = (int)(t / 2);
  float s = 0.f;
  for (int dy = 0; dy < 4; ++dy)
    for (int dx = 0; dx < 4; ++dx)
      s += grads[((long long)(b * 2 + c) * H + (h * 4 + dy)) * H + (w * 4 + dx)];
  float v = (s * (1.f / 16.f)) * gs[0] * ((float)OH / 64.f);
  grid[(((long long)b * OH + h) * OH + w) * 2 + c] = v;
}

// ---------------------- iterated bilinear warp -----------------------------
__global__ void warp_step_k(const float* __restrict__ din, const float* __restrict__ grid,
                            float sign, float* __restrict__ dout, float* __restrict__ s,
                            int B, int C, int H, int W) {
  long long n = (long long)B * C * H * W;
  long long i = (long long)blockIdx.x * blockDim.x + threadIdx.x;
  if (i >= n) return;
  int w = (int)(i % W); long long t = i / W;
  int h = (int)(t % H); t /= H;
  int c = (int)(t % C); int b = (int)(t / C);
  float gx = sign * grid[(((long long)b * H + h) * W + w) * 2 + 0];
  float gy = sign * grid[(((long long)b * H + h) * W + w) * 2 + 1];
  float x = ((gx + 1.f) * (float)W - 1.f) * 0.5f;
  float y = ((gy + 1.f) * (float)H - 1.f) * 0.5f;
  float x0f = floorf(x), y0f = floorf(y);
  int x0 = (int)x0f, y0 = (int)y0f;
  float wx1 = x - x0f, wy1 = y - y0f;
  const float* img = din + ((long long)(b * C + c)) * H * W;
  auto fetch = [&](int xi, int yi) -> float {
    if (xi < 0 || xi > W - 1 || yi < 0 || yi > H - 1) return 0.f;
    return img[(long long)yi * W + xi];
  };
  float v = fetch(x0, y0)         * (1.f - wx1) * (1.f - wy1)
          + fetch(x0 + 1, y0)     * wx1         * (1.f - wy1)
          + fetch(x0, y0 + 1)     * (1.f - wx1) * wy1
          + fetch(x0 + 1, y0 + 1) * wx1         * wy1;
  dout[i] = v;
  s[i]   += v;
}

// ===========================================================================
// Host orchestration
// ===========================================================================
static inline int cdiv(long long a, long long b) { return (int)((a + b - 1) / b); }

extern "C" void kernel_launch(void* const* d_in, const int* in_sizes, int n_in,
                              void* d_out, int out_size, void* d_ws, size_t ws_size,
                              hipStream_t stream) {
  (void)in_sizes; (void)n_in; (void)out_size; (void)ws_size;

  const int B = 2, H0 = 512, W0 = 512;
  const int S512 = 512 * 512, S256 = 256 * 256, S128 = 128 * 128;
  const long long F_ELEMS = (long long)B * 512 * S128;   // output "f" size

  // ---- bump arena in d_ws ----
  char* base = (char*)d_ws; size_t off = 0;
  auto alloc = [&](size_t bytes) -> void* {
    size_t a = (off + 255) & ~(size_t)255; off = a + bytes; return base + a;
  };

  // ---- weight table (setup_inputs dict insertion order) ----
  struct WD { int idx, Co, Ci, K; };
  static const WD wd[] = {
    {1,64,3,7},{2,64,64,3},
    {3,64,64,1},{4,64,64,3},{5,256,64,1},{6,256,64,1},
    {7,64,256,1},{8,64,64,3},{9,256,64,1},
    {10,64,256,1},{11,64,64,3},{12,256,64,1},
    {13,128,256,1},{14,128,128,3},{15,512,128,1},{16,512,256,1},
    {17,128,512,1},{18,128,128,3},{19,512,128,1},
    {20,128,512,1},{21,128,128,3},{22,512,128,1},
    {23,128,512,1},{24,128,128,3},{25,512,128,1},
    {26,128,512,1},{27,128,128,3},{28,512,128,1},
    {29,128,512,1},{30,128,128,3},{31,512,128,1},
    {32,64,512,1},{33,512,640,1},
    {34,128,512,1},{35,128,128,3},{36,512,128,1},{37,512,512,1},
    {38,128,512,1},{39,128,128,3},{40,512,128,1},
    {41,128,512,1},{42,128,128,3},{43,512,128,1},
    {44,128,512,1},{45,128,128,3},{46,512,128,1},
  };
  const int NW = (int)(sizeof(wd) / sizeof(wd[0]));
  const unsigned short* wbf[48]; int wCo[48], wCi[48], wK[48];
  for (int i = 0; i < 48; ++i) { wbf[i] = nullptr; wCo[i] = wCi[i] = wK[i] = 0; }
  for (int i = 0; i < NW; ++i) {
    long long n = (long long)wd[i].Co * wd[i].Ci * wd[i].K * wd[i].K;
    unsigned short* p = (unsigned short*)alloc((size_t)n * 2);
    if (wd[i].idx == 1)   // stem: flat OIHW bf16 for generic kernel
      cvt_bf16_k<<<cdiv(n, 256), 256, 0, stream>>>((const float*)d_in[wd[i].idx], p, n);
    else                  // fast path: [O][KH][KW][I] bf16
      wreorder_k<<<cdiv(n, 256), 256, 0, stream>>>((const float*)d_in[wd[i].idx], p,
                                                   wd[i].Co, wd[i].Ci, wd[i].K);
    wbf[wd[i].idx] = p; wCo[wd[i].idx] = wd[i].Co; wCi[wd[i].idx] = wd[i].Ci; wK[wd[i].idx] = wd[i].K;
  }

  // ---- activation buffers (f32 NCHW; bf16 NHWC) ----
  const size_t f32_big = (size_t)B * 256 * S256 * 4;   // 134 MB (== B*64*S512*4)
  const size_t bf_big  = f32_big / 2;
  float* T_f32 = (float*)alloc(f32_big);
  unsigned short* T_bf = (unsigned short*)alloc(bf_big);
  float* U_f32 = (float*)alloc(f32_big);
  unsigned short* U_bf = (unsigned short*)alloc(bf_big);
  float* SC_f32 = (float*)alloc(f32_big);
  float* Xp_f32[2] = { (float*)alloc(f32_big), (float*)alloc(f32_big) };
  unsigned short* Xp_bf[2] = { (unsigned short*)alloc(bf_big), (unsigned short*)alloc(bf_big) };
  float* grads = (float*)alloc((size_t)B * 2 * S512 * 4);
  unsigned short* finbf = (unsigned short*)alloc((size_t)B * 3 * S512 * 2);
  float* d_mean = (float*)alloc(640 * 4);
  float* d_rstd = (float*)alloc(640 * 4);
  float* gridb  = (float*)alloc((size_t)B * S128 * 2 * 4);
  const size_t d64 = (size_t)B * 64 * S128 * 4;
  float* dA = (float*)alloc(d64); float* dB = (float*)alloc(d64);
  float* dC = (float*)alloc(d64); float* dD = (float*)alloc(d64);
  float* s1 = (float*)alloc(d64); float* s2 = (float*)alloc(d64);
  unsigned short* fbf = (unsigned short*)alloc((size_t)B * 512 * S128 * 2);
  float* catf = (float*)alloc((size_t)B * 640 * S128 * 4);
  unsigned short* catbf = (unsigned short*)alloc((size_t)B * 640 * S128 * 2);

  // ---- launch helpers ----
  auto conv = [&](int widx, const unsigned short* Xb, float* Y,
                  int Hh, int Ww, int stride, int pad) {
    int Kk = wK[widx], Co = wCo[widx], Ci = wCi[widx];
    int OH = (Hh + 2 * pad - Kk) / stride + 1;
    int OW = (Ww + 2 * pad - Kk) / stride + 1;
    int Ntot = B * OH * OW;
    if (widx == 1) {  // stem: generic gather path (Cin==3)
      long long tiles = (long long)((Co + 15) / 16) * ((Ntot + 15) / 16);
      conv_wmma_gen_k<<<cdiv(tiles, 4), 128, 0, stream>>>(wbf[widx], Xb, Y, B, Ci, Hh, Ww,
                                                          Co, Kk, Kk, stride, pad, OH, OW);
    } else {          // fast path: Cin%32==0, Cout%16==0
      long long tiles = (long long)(Co / 16) * ((Ntot + 63) / 64);
      if (pad > 0)
        conv_wmma_fast_k<true><<<cdiv(tiles, 4), 128, 0, stream>>>(
            wbf[widx], Xb, Y, B, Ci, Hh, Ww, Co, Kk, Kk, stride, pad, OH, OW);
      else
        conv_wmma_fast_k<false><<<cdiv(tiles, 4), 128, 0, stream>>>(
            wbf[widx], Xb, Y, B, Ci, Hh, Ww, Co, Kk, Kk, stride, pad, OH, OW);
    }
  };
  auto bnsa = [&](const float* x, int C, int HW, float* y, unsigned short* ybf, int act) {
    bn_stats_k<<<C, 256, 0, stream>>>(x, d_mean, d_rstd, B, C, HW, 1e-5f);
    long long n = (long long)B * C * HW;
    bn_act_k<<<cdiv(n, 256), 256, 0, stream>>>(x, d_mean, d_rstd, y, ybf, C, HW, n, act);
  };
  auto bottleneck = [&](int ic1, int ic2, int ic3, int ids, int stride,
                        const float* xf, const unsigned short* xbf,
                        int Hh, int Ww, float* yf, unsigned short* ybf) {
    int OH = (Hh + 2 - 3) / stride + 1, OW = (Ww + 2 - 3) / stride + 1;
    int planes = wCo[ic1], Cexp = wCo[ic3];
    conv(ic1, xbf, T_f32, Hh, Ww, 1, 0);
    bnsa(T_f32, planes, Hh * Ww, nullptr, T_bf, 1);
    conv(ic2, T_bf, U_f32, Hh, Ww, stride, 1);
    bnsa(U_f32, planes, OH * OW, nullptr, U_bf, 1);
    conv(ic3, U_bf, T_f32, OH, OW, 1, 0);
    bnsa(T_f32, Cexp, OH * OW, T_f32, nullptr, 0);
    const float* res = xf;
    if (ids >= 0) {
      conv(ids, xbf, SC_f32, Hh, Ww, stride, 0);
      bnsa(SC_f32, Cexp, OH * OW, SC_f32, nullptr, 0);
      res = SC_f32;
    }
    long long n = (long long)B * Cexp * OH * OW;
    add_relu_k<<<cdiv(n, 256), 256, 0, stream>>>(T_f32, res, yf, ybf, Cexp, OH * OW, n);
  };

  // ================= forward =================
  {
    long long n = (long long)B * S512;
    sobel_k<<<cdiv(n, 256), 256, 0, stream>>>((const float*)d_in[0], grads, finbf, B, H0, W0);
  }
  {
    long long n = (long long)B * 2 * 128 * 128;
    avg_grid_k<<<cdiv(n, 256), 256, 0, stream>>>(grads, (const float*)d_in[47], gridb, B, 512, 128);
  }
  // conv1 7x7 p3 (generic) -> bn relu
  conv(1, finbf, T_f32, H0, W0, 1, 3);
  bnsa(T_f32, 64, S512, nullptr, T_bf, 1);
  // conv2 3x3 p1 (fast) -> bn relu -> maxpool3s2
  conv(2, T_bf, U_f32, H0, W0, 1, 1);
  bnsa(U_f32, 64, S512, U_f32, nullptr, 1);
  {
    long long n = (long long)B * 64 * S256;
    maxpool_k<<<cdiv(n, 256), 256, 0, stream>>>(U_f32, Xp_f32[0], Xp_bf[0], B, 64, H0, W0);
  }

  // layer2 (3 blocks @256, stride 1)
  int cur = 0;
  bottleneck(3, 4, 5, 6, 1, Xp_f32[cur], Xp_bf[cur], 256, 256, Xp_f32[1 - cur], Xp_bf[1 - cur]); cur ^= 1;
  bottleneck(7, 8, 9, -1, 1, Xp_f32[cur], Xp_bf[cur], 256, 256, Xp_f32[1 - cur], Xp_bf[1 - cur]); cur ^= 1;
  bottleneck(10, 11, 12, -1, 1, Xp_f32[cur], Xp_bf[cur], 256, 256, Xp_f32[1 - cur], Xp_bf[1 - cur]); cur ^= 1;

  // layer3 (6 blocks, first stride 2 -> 128)
  bottleneck(13, 14, 15, 16, 2, Xp_f32[cur], Xp_bf[cur], 256, 256, Xp_f32[1 - cur], Xp_bf[1 - cur]); cur ^= 1;
  bottleneck(17, 18, 19, -1, 1, Xp_f32[cur], Xp_bf[cur], 128, 128, Xp_f32[1 - cur], Xp_bf[1 - cur]); cur ^= 1;
  bottleneck(20, 21, 22, -1, 1, Xp_f32[cur], Xp_bf[cur], 128, 128, Xp_f32[1 - cur], Xp_bf[1 - cur]); cur ^= 1;
  bottleneck(23, 24, 25, -1, 1, Xp_f32[cur], Xp_bf[cur], 128, 128, Xp_f32[1 - cur], Xp_bf[1 - cur]); cur ^= 1;
  bottleneck(26, 27, 28, -1, 1, Xp_f32[cur], Xp_bf[cur], 128, 128, Xp_f32[1 - cur], Xp_bf[1 - cur]); cur ^= 1;
  float* f_out = (float*)d_out;                       // output #1: f (2,512,128,128)
  bottleneck(29, 30, 31, -1, 1, Xp_f32[cur], Xp_bf[cur], 128, 128, f_out, fbf);

  // feature_squeeze (no bn) -> d chains
  conv(32, fbf, dA, 128, 128, 1, 0);
  {
    long long n = (long long)B * 64 * S128;
    copy_k<<<cdiv(n, 256), 256, 0, stream>>>(dA, dC, n);
    fill0_k<<<cdiv(n, 256), 256, 0, stream>>>(s1, n);
    fill0_k<<<cdiv(n, 256), 256, 0, stream>>>(s2, n);
    for (int it = 0; it < 50; ++it) {
      warp_step_k<<<cdiv(n, 256), 256, 0, stream>>>(dA, gridb,  1.f, dB, s1, B, 64, 128, 128);
      warp_step_k<<<cdiv(n, 256), 256, 0, stream>>>(dC, gridb, -1.f, dD, s2, B, 64, 128, 128);
      float* t = dA; dA = dB; dB = t;
      t = dC; dC = dD; dD = t;
    }
  }

  // concat [f, s1, s2] -> tanh(bn) -> merge conv
  {
    long long nf = (long long)B * 512 * S128, ns = (long long)B * 64 * S128;
    copy_into_k<<<cdiv(nf, 256), 256, 0, stream>>>(f_out, catf, B, 512, S128, 640, 0);
    copy_into_k<<<cdiv(ns, 256), 256, 0, stream>>>(s1, catf, B, 64, S128, 640, 512);
    copy_into_k<<<cdiv(ns, 256), 256, 0, stream>>>(s2, catf, B, 64, S128, 640, 576);
  }
  bnsa(catf, 640, S128, nullptr, catbf, 2 /*tanh*/);
  conv(33, catbf, Xp_f32[0], 128, 128, 1, 0);          // merged: (2,512,128,128)
  {
    long long n = (long long)B * 512 * S128;
    nchw2nhwc_bf_k<<<cdiv(n, 256), 256, 0, stream>>>(Xp_f32[0], Xp_bf[0], B, 512, S128);
  }

  // postprocess (4 blocks, first stride 2 -> 64)
  bottleneck(34, 35, 36, 37, 2, Xp_f32[0], Xp_bf[0], 128, 128, Xp_f32[1], Xp_bf[1]);
  bottleneck(38, 39, 40, -1, 1, Xp_f32[1], Xp_bf[1], 64, 64, Xp_f32[0], Xp_bf[0]);
  bottleneck(41, 42, 43, -1, 1, Xp_f32[0], Xp_bf[0], 64, 64, Xp_f32[1], Xp_bf[1]);
  float* out2 = f_out + F_ELEMS;                       // output #2: (2,512,64,64)
  bottleneck(44, 45, 46, -1, 1, Xp_f32[1], Xp_bf[1], 64, 64, out2, nullptr);
}
